// LSTM_84670985273854
// MI455X (gfx1250) — matrix operand order repaired and checked by reading, not compiled, not run on hardware
//
#include <hip/hip_runtime.h>

typedef __attribute__((ext_vector_type(16))) __bf16 v16bf;
typedef __attribute__((ext_vector_type(8)))  float  v8f;
typedef unsigned int v4u __attribute__((ext_vector_type(4)));
typedef int          v4i __attribute__((ext_vector_type(4)));
typedef int          v8i __attribute__((ext_vector_type(8)));

#define B_   64
#define T_   512
#define D_   512
#define H_   512
#define NWG  16     // workgroups; each owns 32 h-columns

// workspace layout (bytes)
#define BPACK_OFF 0                       // 2M bf16  = 4 MB  (W packed in WMMA B-fragment layout)
#define HPACK_OFF (4*1024*1024)           // 2x32768 bf16 = 128 KB (h state, A-fragment layout, double buffered)
#define CNT_OFF   (HPACK_OFF + 128*1024)  // 4 B barrier counter
#define XPACK_OFF (8*1024*1024)           // 16M bf16 = 32 MB (x packed in A-fragment layout per t)
#define HPACK_ELEMS 32768u

#if defined(__has_builtin)
#if __has_builtin(__builtin_amdgcn_tensor_load_to_lds)
#define HAVE_TDM 1
#endif
#endif

__device__ __forceinline__ unsigned short f32_bf16(float f) {
  unsigned u = __float_as_uint(f);
  u += 0x7FFFu + ((u >> 16) & 1u);        // round-to-nearest-even
  return (unsigned short)(u >> 16);
}

#ifdef HAVE_TDM
// 1D TDM copy: n8 units of 8 bytes, global -> LDS. D# per CDNA5 ISA ch.8.
__device__ __forceinline__ void tdm_load_1d(unsigned lds_off, const void* gaddr, unsigned n8) {
  unsigned long long ga = (unsigned long long)(uintptr_t)gaddr;
  v4u g0;
  g0.x = 1u;                                               // count=1 (valid), user mode
  g0.y = lds_off;                                          // lds_addr (bytes)
  g0.z = (unsigned)ga;                                     // global_addr[31:0]
  g0.w = (unsigned)((ga >> 32) & 0x01FFFFFFu) | 0x80000000u; // addr[56:32] | type=2 ("image")
  v8i g1;
  g1[0] = (3 << 16);                                       // data_size = 8 bytes
  g1[1] = (int)((n8 & 0xFFFFu) << 16);                     // tensor_dim0[15:0]
  g1[2] = (int)((n8 >> 16) & 0xFFFFu) | (1 << 16);         // tensor_dim0[31:16] | tensor_dim1 = 1
  g1[3] = (int)((n8 & 0xFFFFu) << 16);                     // tile_dim0 = n8
  g1[4] = 1;                                               // tile_dim1 = 1, tile_dim2 = 0
  g1[5] = (int)n8;                                         // tensor_dim0_stride lo
  g1[6] = 0;
  g1[7] = 0;
  v4i z4 = {0, 0, 0, 0};
#if __clang_major__ >= 23
  v8i z8 = {0, 0, 0, 0, 0, 0, 0, 0};
  __builtin_amdgcn_tensor_load_to_lds(g0, g1, z4, z4, z8, 0);
#else
  __builtin_amdgcn_tensor_load_to_lds(g0, g1, z4, z4, 0);
#endif
}
#endif

// ---- Pack W[1024][2048] fp32 -> bf16 B-fragments -------------------------
// Bpack[(wg*8+wave)][ks][lane][e], wave=(gate*2+nj).
// ISA B layout (32x16 bf16): col N=lane%16,
//                            K = (e&7) + 8*(lane>=16) + 16*(e>=8).
__global__ __launch_bounds__(256) void pack_w_kernel(const float* __restrict__ W,
                                                     unsigned short* __restrict__ bp) {
  unsigned idx  = blockIdx.x * 256u + threadIdx.x;          // < 2^21
  unsigned e    = idx & 15u;
  unsigned lane = (idx >> 4) & 31u;
  unsigned ks   = (idx >> 9) & 31u;
  unsigned wave = (idx >> 14) & 7u;
  unsigned wg   = idx >> 17;
  unsigned gate = wave >> 1, nj = wave & 1u;
  unsigned N    = lane & 15u;
  unsigned K    = (e & 7u) + ((lane >> 4) << 3) + ((e >> 3) << 4);
  unsigned kg   = (ks << 5) + K;
  unsigned ncol = gate * H_ + wg * 32u + nj * 16u + N;
  bp[idx] = f32_bf16(W[(size_t)kg * (4u * H_) + ncol]);
}

// ---- Pack x[B][T][D] fp32 -> bf16 A-fragments per timestep ----------------
// Xpack[t][ks][mi][lane][e]. ISA A layout (16x32 bf16):
// lane%16 = row, K = (e&7) + 8*(lane>=16) + 16*(e>=8).
__global__ __launch_bounds__(256) void pack_x_kernel(const float* __restrict__ x,
                                                     unsigned short* __restrict__ xp) {
  unsigned idx  = blockIdx.x * 256u + threadIdx.x;          // < 2^24
  unsigned e    = idx & 15u;
  unsigned lane = (idx >> 4) & 31u;
  unsigned mi   = (idx >> 9) & 3u;
  unsigned ks   = (idx >> 11) & 15u;
  unsigned t    = idx >> 15;
  unsigned M    = (mi << 4) + (lane & 15u);
  unsigned kk   = (e & 7u) + ((lane >> 4) << 3) + ((e >> 3) << 4);
  unsigned d    = (ks << 5) + kk;
  xp[idx] = f32_bf16(x[((size_t)M * T_ + t) * D_ + d]);
}

__global__ __launch_bounds__(256) void init_kernel(unsigned short* hp, unsigned* cnt) {
  unsigned idx = blockIdx.x * 256u + threadIdx.x;
  if (idx < 2u * HPACK_ELEMS) hp[idx] = 0;                  // h(t=0) = 0
  if (idx == 0) *cnt = 0u;
}

// ---- Persistent LSTM: 16 WGs x 256 threads, all 512 steps in one launch ---
__global__ __launch_bounds__(256) void lstm_persistent(
    const unsigned short* __restrict__ bp,
    const unsigned short* __restrict__ xp,
    unsigned short* hp, unsigned* cnt,
    const float* __restrict__ bias, float* __restrict__ out) {
  __shared__ __align__(128) unsigned short lds_x[2][32768]; // 2 x 64 KB: TDM-staged x_t A-fragments
  __shared__ float zfrag[8192];                             // 8 waves x 4 mi x 256 floats (32 KB)
  const int tid  = threadIdx.x;
  const int wave = tid >> 5, lane = tid & 31;
  const int wg   = blockIdx.x;
  const int g0   = wg * 32;
  // combine-phase ownership: one batch row, 8 consecutive h-columns
  const int row  = tid >> 2;                                // batch index 0..63
  const int c0   = (tid & 3) << 3;
  const int mi_c = row >> 4;
  const int r_c  = row & 7;
  const int lnl_add = (row & 8) << 1;
  const unsigned lds_x_base = (unsigned)(uintptr_t)&lds_x[0][0];

  float c_reg[8];
  float bias_r[4][8];
#pragma unroll
  for (int k = 0; k < 8; ++k) c_reg[k] = 0.f;
#pragma unroll
  for (int g = 0; g < 4; ++g)
#pragma unroll
    for (int k = 0; k < 8; ++k) bias_r[g][k] = bias[g * H_ + g0 + c0 + k];

  const unsigned short* bbase = bp + (((size_t)(wg * 8 + wave)) << 14);  // 16384 elems per (wg,wave)
  const unsigned short* bptr  = bbase + lane * 16;

#ifdef HAVE_TDM
  if (wave == 0)                                            // prefetch x_0 into buffer 0
    tdm_load_1d(lds_x_base, xp, 8192u);
#endif

  for (int t = 0; t < T_; ++t) {
#ifdef HAVE_TDM
    if (wave == 0 && t + 1 < T_)                            // async prefetch x_{t+1}; overlaps phase 0
      tdm_load_1d(lds_x_base + (unsigned)(((t + 1) & 1) << 16),
                  xp + (((size_t)(t + 1)) << 15), 8192u);
#else
    {   // fallback: cooperative staged copy of x_t (no TDM builtin on this toolchain)
      const unsigned short* src = xp + (((size_t)t) << 15);
      unsigned short* dst = &lds_x[t & 1][0];
#pragma unroll
      for (int i = 0; i < 16; ++i)
        *(v8f*)(dst + tid * 128 + i * 8 * 4) = *(const v8f*)(src + tid * 128 + i * 8 * 4);
    }
#endif
    const unsigned short* hrd = hp + ((t & 1) ? HPACK_ELEMS : 0);
    unsigned short*       hwr = hp + ((t & 1) ? 0 : HPACK_ELEMS);

    v8f acc0 = {}, acc1 = {}, acc2 = {}, acc3 = {};
    // ---- phase 0: K = 0..511 (recurrent h, from global/L2) ----
#pragma unroll 2
    for (int ks = 0; ks < 16; ++ks) {
      v16bf bf = *(const v16bf*)(bptr + ks * 512);
      const unsigned short* ab = hrd + ks * 2048 + lane * 16;
      v16bf a0 = *(const v16bf*)(ab);
      v16bf a1 = *(const v16bf*)(ab + 512);
      v16bf a2 = *(const v16bf*)(ab + 1024);
      v16bf a3 = *(const v16bf*)(ab + 1536);
      acc0 = __builtin_amdgcn_wmma_f32_16x16x32_bf16(false, a0, false, bf, (short)0, acc0, false, false);
      acc1 = __builtin_amdgcn_wmma_f32_16x16x32_bf16(false, a1, false, bf, (short)0, acc1, false, false);
      acc2 = __builtin_amdgcn_wmma_f32_16x16x32_bf16(false, a2, false, bf, (short)0, acc2, false, false);
      acc3 = __builtin_amdgcn_wmma_f32_16x16x32_bf16(false, a3, false, bf, (short)0, acc3, false, false);
    }
    // ---- wait for x_t staging DMA, then sync so all waves see LDS ----
#ifdef HAVE_TDM
    if (wave == 0) {
      if (t + 1 < T_) __builtin_amdgcn_s_wait_tensorcnt(1);  // allow x_{t+1} in flight
      else            __builtin_amdgcn_s_wait_tensorcnt(0);
    }
#endif
    __syncthreads();
    // ---- phase 1: K = 512..1023 (x_t, from LDS) ----
    {
      const unsigned short* xl = &lds_x[t & 1][0];
#pragma unroll 2
      for (int ks = 0; ks < 16; ++ks) {
        v16bf bf = *(const v16bf*)(bptr + (ks + 16) * 512);
        const unsigned short* ab = xl + ks * 2048 + lane * 16;
        v16bf a0 = *(const v16bf*)(ab);
        v16bf a1 = *(const v16bf*)(ab + 512);
        v16bf a2 = *(const v16bf*)(ab + 1024);
        v16bf a3 = *(const v16bf*)(ab + 1536);
        acc0 = __builtin_amdgcn_wmma_f32_16x16x32_bf16(false, a0, false, bf, (short)0, acc0, false, false);
        acc1 = __builtin_amdgcn_wmma_f32_16x16x32_bf16(false, a1, false, bf, (short)0, acc1, false, false);
        acc2 = __builtin_amdgcn_wmma_f32_16x16x32_bf16(false, a2, false, bf, (short)0, acc2, false, false);
        acc3 = __builtin_amdgcn_wmma_f32_16x16x32_bf16(false, a3, false, bf, (short)0, acc3, false, false);
      }
    }
    // stage gate fragments through LDS for cross-wave combine
    {
      float* z = zfrag + wave * 1024 + lane * 8;
      *(v8f*)(z      ) = acc0;
      *(v8f*)(z + 256) = acc1;
      *(v8f*)(z + 512) = acc2;
      *(v8f*)(z + 768) = acc3;
    }
    __syncthreads();
    // elementwise LSTM cell: c' = sig(f)*c + sig(i)*cbar ; h' = sig(o)*c'
#pragma unroll
    for (int k = 0; k < 8; ++k) {
      const int col  = c0 + k;
      const int lnl  = (col & 15) + lnl_add;
      const int base = ((col >> 4) * 4 + mi_c) * 256 + lnl * 8 + r_c;
      float f  = zfrag[base       ] + bias_r[0][k];
      float ii = zfrag[base + 2048] + bias_r[1][k];
      float oo = zfrag[base + 4096] + bias_r[2][k];
      float cb = zfrag[base + 6144] + bias_r[3][k];
      float sf = 1.f / (1.f + __expf(-f));
      float si = 1.f / (1.f + __expf(-ii));
      float so = 1.f / (1.f + __expf(-oo));
      float cn = sf * c_reg[k] + si * cb;
      c_reg[k] = cn;
      float hv = so * cn;
      out[((size_t)row * T_ + t) * H_ + g0 + col] = hv;
      // scatter h' back into A-fragment layout (next step's K input); ksH == wg
      const int lw = (row & 15) + ((col & 8) << 1);
      const int ew = (col & 7) + ((col >> 4) << 3);
      hwr[((wg * 4 + mi_c) * 32 + lw) * 16 + ew] = f32_bf16(hv);
    }
    // ---- grid-wide step barrier ----
    __threadfence();                 // release h' stores to agent scope
    __syncthreads();
    if (tid == 0) {
      __hip_atomic_fetch_add(cnt, 1u, __ATOMIC_RELEASE, __HIP_MEMORY_SCOPE_AGENT);
      const unsigned target = (unsigned)(t + 1) * NWG;
      while (__hip_atomic_load(cnt, __ATOMIC_ACQUIRE, __HIP_MEMORY_SCOPE_AGENT) < target)
        __builtin_amdgcn_s_sleep(2);
    }
#if defined(__has_builtin)
#if __has_builtin(__builtin_amdgcn_s_cluster_barrier)
    __builtin_amdgcn_s_cluster_barrier();  // s_barrier_signal/-wait -3; S_NOP if not cluster-dispatched
#endif
#endif
    __syncthreads();
    __builtin_amdgcn_fence(__ATOMIC_ACQUIRE, "agent");  // invalidate stale WGP$ lines before next K loop
  }
}

extern "C" void kernel_launch(void* const* d_in, const int* in_sizes, int n_in,
                              void* d_out, int out_size, void* d_ws, size_t ws_size,
                              hipStream_t stream) {
  (void)in_sizes; (void)n_in; (void)out_size; (void)ws_size;
  const float* x = (const float*)d_in[0];
  const float* W = (const float*)d_in[1];
  const float* b = (const float*)d_in[2];
  float* out = (float*)d_out;
  char* ws = (char*)d_ws;
  unsigned short* bpack = (unsigned short*)(ws + BPACK_OFF);
  unsigned short* hpack = (unsigned short*)(ws + HPACK_OFF);
  unsigned*       cnt   = (unsigned*)(ws + CNT_OFF);
  unsigned short* xpack = (unsigned short*)(ws + XPACK_OFF);

  pack_w_kernel<<<dim3(8192),  dim3(256), 0, stream>>>(W, bpack);
  pack_x_kernel<<<dim3(65536), dim3(256), 0, stream>>>(x, xpack);
  init_kernel  <<<dim3(256),   dim3(256), 0, stream>>>(hpack, cnt);
  lstm_persistent<<<dim3(NWG), dim3(256), 0, stream>>>(bpack, xpack, hpack, cnt, b, out);
}
